// CrossAttention_20005957664900
// MI455X (gfx1250) — matrix operand order repaired and checked by reading, not compiled, not run on hardware
//
#include <hip/hip_runtime.h>

// ---------------------------------------------------------------------------
// CrossAttention forward for MI455X (gfx1250), bf16 WMMA pipeline.
//   1. transpose+convert weights to bf16 W^T (N x K)
//   2. Q = x @ Wq, K = ctx @ Wk (bf16), V = ctx @ Wv stored transposed
//   3. flash-attention (online softmax) with v_wmma_f32_16x16x32_bf16
//   4. out = O @ Wo + bo (fp32 output)
// ---------------------------------------------------------------------------

#define HEADS 16
#define DIM_HEAD 64

typedef __bf16 bf16;
typedef __attribute__((ext_vector_type(16))) __bf16 v16bf;
typedef __attribute__((ext_vector_type(8)))  __bf16 bf16x8;   // 16 bytes
typedef __attribute__((ext_vector_type(8)))  float  v8f;
typedef __attribute__((ext_vector_type(4)))  float  f32x4;

union AF { v16bf v; bf16x8 h[2]; };

// ---- WMMA fragment loaders (layouts per CDNA5 ISA 7.12.2, wave32) ----------

// A-matrix 16x32 bf16: lane L(0-15) row M=L, K base 0; lane L+16 same row, K base 8.
// element i(0..7) = K base+i ; element i(8..15) = K base+16+(i-8).
__device__ __forceinline__ v16bf frag_a_bf16(const bf16* A, int lda, int row0,
                                             int k0, int lane) {
  int r  = row0 + (lane & 15);
  int kb = k0 + ((lane >> 4) << 3);
  const bf16* p = A + (size_t)r * lda + kb;
  AF f;
  f.h[0] = *(const bf16x8*)(p);
  f.h[1] = *(const bf16x8*)(p + 16);
  return f.v;
}

// Same A layout but source is fp32 (convert to bf16 while loading).
__device__ __forceinline__ v16bf frag_a_f32(const float* A, int lda, int row0,
                                            int k0, int lane) {
  int r  = row0 + (lane & 15);
  int kb = k0 + ((lane >> 4) << 3);
  const float* p = A + (size_t)r * lda + kb;
  f32x4 a0 = *(const f32x4*)(p);
  f32x4 a1 = *(const f32x4*)(p + 4);
  f32x4 a2 = *(const f32x4*)(p + 16);
  f32x4 a3 = *(const f32x4*)(p + 20);
  v16bf v;
#pragma unroll
  for (int i = 0; i < 4; ++i) {
    v[i]      = (bf16)a0[i];
    v[4 + i]  = (bf16)a1[i];
    v[8 + i]  = (bf16)a2[i];
    v[12 + i] = (bf16)a3[i];
  }
  return v;
}

// B-matrix 32x16 bf16, read from Bt stored as (N x K) row-major (Bt[n][k]=B[k][n]):
// lane n(0-15) col N=n, K=k0..k0+7,k0+16..k0+23 ; lane n+16 same col, K=k0+8..k0+15,k0+24..k0+31.
__device__ __forceinline__ v16bf frag_b_nk(const bf16* Bt, int ldb, int n0,
                                           int k0, int lane) {
  int n  = n0 + (lane & 15);
  int kb = k0 + ((lane >> 4) << 3);
  const bf16* p = Bt + (size_t)n * ldb + kb;
  AF f;
  f.h[0] = *(const bf16x8*)(p);
  f.h[1] = *(const bf16x8*)(p + 16);
  return f.v;
}

__device__ __forceinline__ v8f wmma_bf16(v16bf a, v16bf b, v8f c) {
  // (neg_a, A, neg_b, B, c_mod, C, reuse_a, reuse_b)
  return __builtin_amdgcn_wmma_f32_16x16x32_bf16(false, a, false, b, (short)0,
                                                 c, false, false);
}

// ---- weight transpose + fp32->bf16 ----------------------------------------
// W: (K x N) fp32 row-major  ->  Wt: (N x K) bf16 row-major
__global__ __launch_bounds__(256) void transpose_to_bf16(
    const float* __restrict__ W, bf16* __restrict__ Wt, int K, int N) {
  int idx = blockIdx.x * blockDim.x + threadIdx.x;
  if (idx >= N * K) return;
  int n = idx / K;
  int k = idx - n * K;
  Wt[idx] = (bf16)W[(size_t)k * N + n];
}

// ---- generic WMMA GEMM: C(MxN) = A(MxK) * Bt^T, one wave -> 16x64 strip ----
// AMODE: 0 = A fp32, 1 = A bf16
// OMODE: 0 = bf16 row-major, 1 = bf16 V-transpose (b,h,d,s), 2 = fp32 + bias
template <int AMODE, int OMODE>
__global__ __launch_bounds__(256) void gemm_wmma_16x64(
    const void* __restrict__ Ap, const bf16* __restrict__ Bt,
    void* __restrict__ Cp, const float* __restrict__ bias,
    int M, int N, int K) {
  const int lane = threadIdx.x & 31;
  const int gw = blockIdx.x * (blockDim.x >> 5) + (threadIdx.x >> 5);
  const int ntiles = N >> 6;
  const int mt = gw / ntiles;
  const int nt = gw - mt * ntiles;
  if (mt * 16 >= M) return;
  const int row0 = mt * 16;
  const int n0 = nt * 64;

  v8f acc[4] = {};
  for (int k0 = 0; k0 < K; k0 += 32) {
    v16bf a = (AMODE == 0) ? frag_a_f32((const float*)Ap, K, row0, k0, lane)
                           : frag_a_bf16((const bf16*)Ap, K, row0, k0, lane);
    if (k0 + 32 < K) {  // global_prefetch_b8 for the next A slab
      size_t eb = (size_t)(row0 + (lane & 15)) * K + k0 + 32;
      __builtin_prefetch((const char*)Ap + eb * (AMODE == 0 ? 4 : 2), 0, 1);
    }
#pragma unroll
    for (int t = 0; t < 4; ++t) {
      v16bf b = frag_b_nk(Bt, K, n0 + t * 16, k0, lane);
      acc[t] = wmma_bf16(a, b, acc[t]);
    }
  }

  // C/D layout: VGPR j, lane c(0-15): row=row0+j, col=n0+c ; lane c+16: row+8.
  const int half8 = (lane >> 4) << 3;
#pragma unroll
  for (int t = 0; t < 4; ++t) {
#pragma unroll
    for (int j = 0; j < 8; ++j) {
      int row = row0 + j + half8;
      int col = n0 + t * 16 + (lane & 15);
      float val = acc[t][j];
      if (OMODE == 0) {
        ((bf16*)Cp)[(size_t)row * N + col] = (bf16)val;
      } else if (OMODE == 1) {
        // row = b*1024 + s ; col = h*64 + d ; store Vt[b][h][d][s]
        int b = row >> 10, s = row & 1023;
        int h = col >> 6, d = col & 63;
        ((bf16*)Cp)[(((size_t)(b * HEADS + h)) * DIM_HEAD + d) * 1024 + s] =
            (bf16)val;
      } else {
        ((float*)Cp)[(size_t)row * N + col] = val + bias[col];
      }
    }
  }
}

// ---- flash attention: one wave per (b, h, 16 query rows) -------------------
__global__ __launch_bounds__(256) void attn_fwd(
    const bf16* __restrict__ Q,   // (B, SQ, 1024)
    const bf16* __restrict__ Kb,  // (B, SKV, 1024)
    const bf16* __restrict__ Vt,  // (B, H, 64, SKV)
    bf16* __restrict__ O) {       // (B, SQ, 1024)
  const int SQ = 4096, SKV = 1024, ID = 1024;
  __shared__ __align__(16) bf16 Plds[8][16 * 32];  // 1KB per wave

  const int lane = threadIdx.x & 31;
  const int w = threadIdx.x >> 5;
  const int gw = blockIdx.x * 8 + w;      // grid sized exactly: no guard
  const int qt = gw & 255;                // SQ/16 tiles per (b,h)
  const int bh = gw >> 8;                 // 0..63
  const int h = bh & 15, b = bh >> 4;
  const int row0 = qt * 16;

  const bf16* Qbase = Q + ((size_t)b * SQ) * ID + h * DIM_HEAD;
  const bf16* Kbase = Kb + ((size_t)b * SKV) * ID + h * DIM_HEAD;
  const bf16* Vbase = Vt + ((size_t)(b * HEADS + h)) * DIM_HEAD * SKV;

  // Q fragment for the whole row-block (D=64 -> two k-steps), kept resident.
  v16bf qf0 = frag_a_bf16(Qbase, ID, row0, 0, lane);
  v16bf qf1 = frag_a_bf16(Qbase, ID, row0, 32, lane);

  float m[8], l[8];
  v8f o0 = {}, o1 = {}, o2 = {}, o3 = {};
#pragma unroll
  for (int j = 0; j < 8; ++j) { m[j] = -1e30f; l[j] = 0.f; }

  bf16* pl = &Plds[w][0];
  const float scale = 0.125f;  // 64^-0.5
  const int half8 = (lane >> 4) << 3;

  for (int kv = 0; kv < SKV; kv += 32) {
    // ---- S = Q * K^T for a 16x32 chunk (B-frag = rows of K, contiguous) ----
    const bf16* Kchunk = Kbase + (size_t)kv * ID;
    v8f s0 = {}, s1 = {};
    s0 = wmma_bf16(qf0, frag_b_nk(Kchunk, ID, 0, 0, lane), s0);
    s0 = wmma_bf16(qf1, frag_b_nk(Kchunk, ID, 0, 32, lane), s0);
    s1 = wmma_bf16(qf0, frag_b_nk(Kchunk, ID, 16, 0, lane), s1);
    s1 = wmma_bf16(qf1, frag_b_nk(Kchunk, ID, 16, 32, lane), s1);

    // ---- online softmax: each lane holds rows {j+half8}, cols lane&15 ------
#pragma unroll
    for (int j = 0; j < 8; ++j) {
      float v0 = s0[j] * scale, v1 = s1[j] * scale;
      float mj = fmaxf(v0, v1);
#pragma unroll
      for (int off = 1; off < 16; off <<= 1)
        mj = fmaxf(mj, __shfl_xor(mj, off, 32));
      float mn = fmaxf(m[j], mj);
      float alpha = __expf(m[j] - mn);
      float p0 = __expf(v0 - mn), p1 = __expf(v1 - mn);
      float rs = p0 + p1;
#pragma unroll
      for (int off = 1; off < 16; off <<= 1) rs += __shfl_xor(rs, off, 32);
      l[j] = l[j] * alpha + rs;
      m[j] = mn;
      o0[j] *= alpha; o1[j] *= alpha; o2[j] *= alpha; o3[j] *= alpha;
      s0[j] = p0; s1[j] = p1;
    }

    // ---- C-layout P -> LDS (row-major 16x32 bf16) -> reload as A-fragment --
#pragma unroll
    for (int j = 0; j < 8; ++j) {
      int r = j + half8;
      pl[r * 32 + (lane & 15)]      = (bf16)s0[j];
      pl[r * 32 + 16 + (lane & 15)] = (bf16)s1[j];
    }
    __syncthreads();
    v16bf pf = frag_a_bf16(pl, 32, 0, 0, lane);
    __syncthreads();

    // ---- O += P * V  (B-frag = rows of Vt = columns of V, contiguous) ------
    o0 = wmma_bf16(pf, frag_b_nk(Vbase, SKV, 0,  kv, lane), o0);
    o1 = wmma_bf16(pf, frag_b_nk(Vbase, SKV, 16, kv, lane), o1);
    o2 = wmma_bf16(pf, frag_b_nk(Vbase, SKV, 32, kv, lane), o2);
    o3 = wmma_bf16(pf, frag_b_nk(Vbase, SKV, 48, kv, lane), o3);
  }

  // ---- epilogue: O / l, store bf16 into (b, sq, h*64 + d) -------------------
  bf16* Ob = O + ((size_t)b * SQ + row0) * ID + h * DIM_HEAD;
#pragma unroll
  for (int j = 0; j < 8; ++j) {
    float inv = 1.0f / l[j];
    int r = j + half8;
    Ob[(size_t)r * ID + 0  + (lane & 15)] = (bf16)(o0[j] * inv);
    Ob[(size_t)r * ID + 16 + (lane & 15)] = (bf16)(o1[j] * inv);
    Ob[(size_t)r * ID + 32 + (lane & 15)] = (bf16)(o2[j] * inv);
    Ob[(size_t)r * ID + 48 + (lane & 15)] = (bf16)(o3[j] * inv);
  }
}

// ---------------------------------------------------------------------------
extern "C" void kernel_launch(void* const* d_in, const int* in_sizes, int n_in,
                              void* d_out, int out_size, void* d_ws,
                              size_t ws_size, hipStream_t stream) {
  const float* x   = (const float*)d_in[0];  // (4,4096,1024)
  const float* ctx = (const float*)d_in[1];  // (4,1024,768)
  const float* Wq  = (const float*)d_in[2];  // (1024,1024)
  const float* Wk  = (const float*)d_in[3];  // (768,1024)
  const float* Wv  = (const float*)d_in[4];  // (768,1024)
  const float* Wo  = (const float*)d_in[5];  // (1024,1024)
  const float* bo  = (const float*)d_in[6];  // (1024,)

  const int B = 4, SQ = 4096, SKV = 1024, QD = 1024, CD = 768, ID = 1024;

  char* ws = (char*)d_ws;
  size_t off = 0;
  auto alloc = [&](size_t bytes) {
    void* p = ws + off;
    off += (bytes + 255) & ~(size_t)255;
    return p;
  };
  bf16* WqT = (bf16*)alloc((size_t)ID * QD * 2);      // (ID x QD)
  bf16* WkT = (bf16*)alloc((size_t)ID * CD * 2);      // (ID x CD)
  bf16* WvT = (bf16*)alloc((size_t)ID * CD * 2);      // (ID x CD)
  bf16* WoT = (bf16*)alloc((size_t)QD * ID * 2);      // (QD x ID)
  bf16* Qb  = (bf16*)alloc((size_t)B * SQ * ID * 2);  // (B,SQ,ID)
  bf16* Kbf = (bf16*)alloc((size_t)B * SKV * ID * 2); // (B,SKV,ID)
  bf16* Vtb = (bf16*)alloc((size_t)B * ID * SKV * 2); // (B,H,64,SKV)
  bf16* Ob  = (bf16*)alloc((size_t)B * SQ * ID * 2);  // (B,SQ,ID)

  auto blocks = [](long n, int per) {
    return dim3((unsigned)((n + per - 1) / per));
  };

  // 1) weights -> bf16 transposed
  transpose_to_bf16<<<blocks((long)QD * ID, 256), 256, 0, stream>>>(Wq, WqT, QD, ID);
  transpose_to_bf16<<<blocks((long)CD * ID, 256), 256, 0, stream>>>(Wk, WkT, CD, ID);
  transpose_to_bf16<<<blocks((long)CD * ID, 256), 256, 0, stream>>>(Wv, WvT, CD, ID);
  transpose_to_bf16<<<blocks((long)ID * QD, 256), 256, 0, stream>>>(Wo, WoT, ID, QD);

  // 2) projections (8 waves / block)
  {
    long waves = (long)(B * SQ / 16) * (ID / 64);
    gemm_wmma_16x64<0, 0><<<blocks(waves, 8), 256, 0, stream>>>(
        x, WqT, Qb, nullptr, B * SQ, ID, QD);
  }
  {
    long waves = (long)(B * SKV / 16) * (ID / 64);
    gemm_wmma_16x64<0, 0><<<blocks(waves, 8), 256, 0, stream>>>(
        ctx, WkT, Kbf, nullptr, B * SKV, ID, CD);
    gemm_wmma_16x64<0, 1><<<blocks(waves, 8), 256, 0, stream>>>(
        ctx, WvT, Vtb, nullptr, B * SKV, ID, CD);
  }

  // 3) attention: B*H*(SQ/16) = 16384 waves exactly -> 2048 blocks
  {
    long waves = (long)B * HEADS * (SQ / 16);
    attn_fwd<<<blocks(waves, 8), 256, 0, stream>>>(Qb, Kbf, Vtb, Ob);
  }

  // 4) output projection + bias (fp32 out)
  {
    long waves = (long)(B * SQ / 16) * (QD / 64);
    gemm_wmma_16x64<1, 2><<<blocks(waves, 8), 256, 0, stream>>>(
        Ob, WoT, d_out, bo, B * SQ, QD, ID);
  }
}
